// GPTBigCodeAttention_70033736729077
// MI455X (gfx1250) — matrix operand order repaired and checked by reading, not compiled, not run on hardware
//
#include <hip/hip_runtime.h>
#include <hip/hip_bf16.h>

typedef __attribute__((ext_vector_type(16))) __bf16 v16bf;
typedef __attribute__((ext_vector_type(8)))  float  v8f;
typedef __attribute__((ext_vector_type(4)))  int    v4i;

typedef __attribute__((address_space(1))) v4i gv4i;   // global int4
typedef __attribute__((address_space(3))) v4i lv4i;   // LDS int4

union Frag { v16bf v; unsigned int u[8]; };

__device__ __forceinline__ unsigned short f2bf(float f) {
    unsigned int u = __float_as_uint(f);
    unsigned int r = u + 0x7FFFu + ((u >> 16) & 1u);   // RNE
    return (unsigned short)(r >> 16);
}

// 16-byte async global->LDS copy (gfx1250 GLOBAL_LOAD_ASYNC_TO_LDS_B128,
// tracked by ASYNCcnt; per-lane global and LDS addresses).
__device__ __forceinline__ void async_b128(const unsigned short* g, unsigned short* l) {
    __builtin_amdgcn_global_load_async_to_lds_b128(
        (gv4i*)(unsigned short*)g,
        (lv4i*)l,
        0, 0);
}

// ---------------------------------------------------------------------------
// fp32 -> bf16 elementwise convert (vectorized x4)
// ---------------------------------------------------------------------------
__global__ __launch_bounds__(256) void f32_to_bf16_kernel(
    const float* __restrict__ in, unsigned short* __restrict__ out, int n)
{
    int i = (blockIdx.x * 256 + threadIdx.x) * 4;
    if (i + 3 < n) {
        float4 f = *(const float4*)(in + i);
        out[i + 0] = f2bf(f.x);
        out[i + 1] = f2bf(f.y);
        out[i + 2] = f2bf(f.z);
        out[i + 3] = f2bf(f.w);
    }
}

// ---------------------------------------------------------------------------
// W[K][N] fp32 -> Wt[N][K] bf16 (tiled transpose through LDS)
// ---------------------------------------------------------------------------
__global__ __launch_bounds__(256) void transpose_f32_to_bf16(
    const float* __restrict__ W, unsigned short* __restrict__ Wt, int K, int N)
{
    __shared__ float tile[32][33];
    int n0 = blockIdx.x * 32, k0 = blockIdx.y * 32;
    int tx = threadIdx.x & 31, ty = threadIdx.x >> 5;   // 32 x 8
    #pragma unroll
    for (int i = 0; i < 32; i += 8)
        tile[ty + i][tx] = W[(size_t)(k0 + ty + i) * N + (n0 + tx)];
    __syncthreads();
    #pragma unroll
    for (int i = 0; i < 32; i += 8)
        Wt[(size_t)(n0 + ty + i) * K + (k0 + tx)] = f2bf(tile[tx][ty + i]);
}

// ---------------------------------------------------------------------------
// V columns of qkv_bf [B*S][2304] -> vt [B][128][S] (bf16 transpose)
// ---------------------------------------------------------------------------
__global__ __launch_bounds__(256) void transpose_v_bf16(
    const unsigned short* __restrict__ qkv_bf, unsigned short* __restrict__ vt, int S)
{
    __shared__ unsigned short tile[32][33];
    int s0 = blockIdx.x * 32, d0 = blockIdx.y * 32, b = blockIdx.z;
    int tx = threadIdx.x & 31, ty = threadIdx.x >> 5;
    #pragma unroll
    for (int i = 0; i < 32; i += 8)
        tile[ty + i][tx] = qkv_bf[(size_t)(b * S + s0 + ty + i) * 2304 + 2176 + d0 + tx];
    __syncthreads();
    #pragma unroll
    for (int i = 0; i < 32; i += 8)
        vt[((size_t)b * 128 + d0 + ty + i) * S + (s0 + tx)] = tile[tx][ty + i];
}

// ---------------------------------------------------------------------------
// C = A[M][K](bf16) * Bt[N][K](bf16)^T + bias.  Output: fp32 Cf and/or bf16 Cbf.
// 128x128 workgroup tile, 8 waves x (16 x 128).  Double-buffered LDS staged
// with async global->LDS copies; last K-step peeled so the steady-state loop
// is straight-line: prefetch -> s_wait_asynccnt 4 -> barrier -> WMMA chain.
// ---------------------------------------------------------------------------
__global__ __launch_bounds__(256) void gemm_bf16_wmma(
    const unsigned short* __restrict__ A,   // [M][K]
    const unsigned short* __restrict__ Bt,  // [N][K]
    const float* __restrict__ bias,         // [N]
    float* __restrict__ Cf,                 // [M][N] fp32 (nullable)
    unsigned short* __restrict__ Cbf,       // [M][N] bf16 (nullable)
    int M, int N, int K)
{
    __shared__ __align__(16) unsigned short As[2][128][40];
    __shared__ __align__(16) unsigned short Bs[2][128][40];

    const int tid  = threadIdx.x;
    const int lane = tid & 31;
    const int w    = tid >> 5;
    const int m0   = blockIdx.y * 128;
    const int n0   = blockIdx.x * 128;
    const int row  = lane & 15;
    const int g    = lane >> 4;
    const int lr   = tid >> 1;            // staging row 0..127
    const int lc   = (tid & 1) * 16;      // staging col half

    const unsigned short* gA = A  + (size_t)(m0 + lr) * K + lc;
    const unsigned short* gB = Bt + (size_t)(n0 + lr) * K + lc;

    const v8f vzero = {0.f, 0.f, 0.f, 0.f, 0.f, 0.f, 0.f, 0.f};
    v8f acc[8];
    #pragma unroll
    for (int ns = 0; ns < 8; ++ns) acc[ns] = vzero;

    auto stage = [&](int k0, int bb) {
        async_b128(gA + k0,     &As[bb][lr][lc]);
        async_b128(gA + k0 + 8, &As[bb][lr][lc + 8]);
        async_b128(gB + k0,     &Bs[bb][lr][lc]);
        async_b128(gB + k0 + 8, &Bs[bb][lr][lc + 8]);
    };

    auto compute_tile = [&](int bb) {
        // A fragment (16x32 bf16, documented per-lane K striping)
        Frag a;
        const unsigned short* ar = &As[bb][w * 16 + row][0];
        #pragma unroll
        for (int v = 0; v < 4; ++v)
            a.u[v] = *(const unsigned int*)(ar + g * 8 + 2 * v);
        #pragma unroll
        for (int v = 4; v < 8; ++v)
            a.u[v] = *(const unsigned int*)(ar + 16 + g * 8 + 2 * (v - 4));
        // B fragments + WMMA chain
        Frag bfr[8];
        #pragma unroll
        for (int ns = 0; ns < 8; ++ns) {
            const unsigned short* br = &Bs[bb][ns * 16 + row][0];
            #pragma unroll
            for (int v = 0; v < 8; ++v)
                bfr[ns].u[v] = *(const unsigned int*)(br + (v >> 2) * 16 + g * 8 + 2 * (v & 3));
        }
        #pragma unroll
        for (int ns = 0; ns < 8; ++ns)
            acc[ns] = __builtin_amdgcn_wmma_f32_16x16x32_bf16(
                false, a.v, false, bfr[ns].v, (short)0, acc[ns], false, false);
    };

    stage(0, 0);                          // prologue
    int buf = 0;
    for (int k0 = 0; k0 < K - 32; k0 += 32) {
        stage(k0 + 32, buf ^ 1);          // prefetch next K-tile
        asm volatile("s_wait_asynccnt 0x4" ::: "memory");   // current tile landed
        __syncthreads();
        compute_tile(buf);
        __syncthreads();                  // buf free for refill
        buf ^= 1;
    }
    asm volatile("s_wait_asynccnt 0x0" ::: "memory");       // peeled last tile
    __syncthreads();
    compute_tile(buf);

    #pragma unroll
    for (int ns = 0; ns < 8; ++ns) {
        int col = n0 + ns * 16 + row;
        float bs = bias[col];
        #pragma unroll
        for (int v = 0; v < 8; ++v) {
            int r = m0 + w * 16 + v + 8 * g;
            float val = acc[ns][v] + bs;
            if (Cf)  Cf [(size_t)r * N + col] = val;
            if (Cbf) Cbf[(size_t)r * N + col] = f2bf(val);
        }
    }
}

// ---------------------------------------------------------------------------
// Flash attention (MQA, causal), all-bf16 staging via async global->LDS with
// double-buffered K/V tiles (prefetch kb+1 under compute of kb).
//   qkv_bf [B*S][2304]: Q = [:, h*128:+128], K = [:, 2048:2176]
//   vt     [B][128][S]: V transposed
// 128 threads = 4 waves; wave: 16 query rows; WG: 64 rows of one (b,h).
// ---------------------------------------------------------------------------
__global__ __launch_bounds__(128) void attn_flash_wmma(
    const unsigned short* __restrict__ qkv_bf,
    const unsigned short* __restrict__ vt,
    unsigned short* __restrict__ out_bf,   // [B*S][2048]
    int Sdim)
{
    __shared__ __align__(16) unsigned short Qs[64][136];
    __shared__ __align__(16) unsigned short Ks[2][32][136];
    __shared__ __align__(16) unsigned short Vt[2][128][40];   // [d][key]
    __shared__ __align__(16) unsigned short Ps[4][16][40];    // per-wave P scratch

    const int tid  = threadIdx.x;
    const int lane = tid & 31;
    const int w    = tid >> 5;
    const int row  = lane & 15;
    const int g    = lane >> 4;

    const int qb = blockIdx.x;
    const int b  = blockIdx.y >> 4;
    const int h  = blockIdx.y & 15;
    const float scale = 0.08838834764831845f;   // 1/sqrt(128)
    const size_t rowQ0 = (size_t)(b * Sdim + qb * 64);

    const int kr = tid >> 2, kc = (tid & 3) * 32;
    const unsigned short* srcK0 = qkv_bf + ((size_t)b * Sdim + kr) * 2304 + 2048 + kc;
    const unsigned short* srcV0 = vt + ((size_t)b * 128 + tid) * Sdim;  // d = tid

    auto stageKV = [&](int kb, int bb) {        // 8 async ops per wave
        const unsigned short* sk = srcK0 + (size_t)kb * 32 * 2304;
        #pragma unroll
        for (int j = 0; j < 32; j += 8)
            async_b128(sk + j, &Ks[bb][kr][kc + j]);
        const unsigned short* sv = srcV0 + kb * 32;
        #pragma unroll
        for (int j = 0; j < 32; j += 8)
            async_b128(sv + j, &Vt[bb][tid][j]);
    };

    {   // stage Q (raw bf16 async copy, 8 ops/wave)
        int r  = tid >> 1;
        int ch = (tid & 1) * 64;
        const unsigned short* src = qkv_bf + (rowQ0 + r) * 2304 + h * 128 + ch;
        #pragma unroll
        for (int j = 0; j < 64; j += 8)
            async_b128(src + j, &Qs[r][ch + j]);
    }
    stageKV(0, 0);                              // prefetch first K/V tile
    asm volatile("s_wait_asynccnt 0x8" ::: "memory");   // Q landed (tile 0 in flight)
    __syncthreads();

    // preload Q A-fragments (constant across key blocks)
    Frag qf[4];
    #pragma unroll
    for (int d = 0; d < 4; ++d) {
        const unsigned short* ar = &Qs[w * 16 + row][d * 32];
        #pragma unroll
        for (int v = 0; v < 4; ++v)
            qf[d].u[v] = *(const unsigned int*)(ar + g * 8 + 2 * v);
        #pragma unroll
        for (int v = 4; v < 8; ++v)
            qf[d].u[v] = *(const unsigned int*)(ar + 16 + g * 8 + 2 * (v - 4));
    }

    const v8f vzero = {0.f, 0.f, 0.f, 0.f, 0.f, 0.f, 0.f, 0.f};
    v8f acc_o[8];
    #pragma unroll
    for (int d = 0; d < 8; ++d) acc_o[d] = vzero;
    float mrow[8], lrow[8];
    #pragma unroll
    for (int v = 0; v < 8; ++v) { mrow[v] = -1e30f; lrow[v] = 0.f; }

    const int kbmax = qb * 2 + 1;
    int buf = 0;
    for (int kb = 0; kb <= kbmax; ++kb) {
        if (kb < kbmax) {
            stageKV(kb + 1, buf ^ 1);           // prefetch next K/V tile
            asm volatile("s_wait_asynccnt 0x8" ::: "memory");   // current tile landed
        } else {
            asm volatile("s_wait_asynccnt 0x0" ::: "memory");
        }
        __syncthreads();

        // scores: gather K fragments, then chained WMMAs (K-dim = 128)
        v8f sc[2];
        #pragma unroll
        for (int sub = 0; sub < 2; ++sub) {
            Frag kf[4];
            #pragma unroll
            for (int d = 0; d < 4; ++d) {
                const unsigned short* br = &Ks[buf][sub * 16 + row][d * 32];
                #pragma unroll
                for (int v = 0; v < 8; ++v)
                    kf[d].u[v] = *(const unsigned int*)(br + (v >> 2) * 16 + g * 8 + 2 * (v & 3));
            }
            v8f s = vzero;
            #pragma unroll
            for (int d = 0; d < 4; ++d)
                s = __builtin_amdgcn_wmma_f32_16x16x32_bf16(
                    false, qf[d].v, false, kf[d].v, (short)0, s, false, false);
            sc[sub] = s;
        }

        // scale + causal mask + online softmax
        const int kbase = kb * 32;
        const int qbase = qb * 64 + w * 16;
        #pragma unroll
        for (int v = 0; v < 8; ++v) {
            int q = qbase + v + 8 * g;
            float s0 = sc[0][v] * scale; if (kbase + row > q)      s0 = -1e30f;
            float s1 = sc[1][v] * scale; if (kbase + 16 + row > q) s1 = -1e30f;
            float t = fmaxf(s0, s1);
            t = fmaxf(t, __shfl_xor(t, 1, 32));
            t = fmaxf(t, __shfl_xor(t, 2, 32));
            t = fmaxf(t, __shfl_xor(t, 4, 32));
            t = fmaxf(t, __shfl_xor(t, 8, 32));
            float mnew = fmaxf(mrow[v], t);
            float corr = __expf(mrow[v] - mnew);
            float p0 = __expf(s0 - mnew);
            float p1 = __expf(s1 - mnew);
            sc[0][v] = p0; sc[1][v] = p1;
            float rs = p0 + p1;
            rs += __shfl_xor(rs, 1, 32);
            rs += __shfl_xor(rs, 2, 32);
            rs += __shfl_xor(rs, 4, 32);
            rs += __shfl_xor(rs, 8, 32);
            lrow[v] = lrow[v] * corr + rs;
            mrow[v] = mnew;
            #pragma unroll
            for (int d = 0; d < 8; ++d) acc_o[d][v] *= corr;
        }

        // P: C-layout -> LDS -> A-layout fragment (wave-private scratch)
        #pragma unroll
        for (int v = 0; v < 8; ++v) {
            Ps[w][v + 8 * g][row]      = f2bf(sc[0][v]);
            Ps[w][v + 8 * g][16 + row] = f2bf(sc[1][v]);
        }
        asm volatile("s_wait_dscnt 0x0" ::: "memory");
        Frag pf;
        {
            const unsigned short* pr = &Ps[w][row][0];
            #pragma unroll
            for (int v = 0; v < 4; ++v)
                pf.u[v] = *(const unsigned int*)(pr + g * 8 + 2 * v);
            #pragma unroll
            for (int v = 4; v < 8; ++v)
                pf.u[v] = *(const unsigned int*)(pr + 16 + g * 8 + 2 * (v - 4));
        }

        // O += P @ V : gather all 8 V fragments, then WMMA chain
        Frag vf[8];
        #pragma unroll
        for (int dsub = 0; dsub < 8; ++dsub) {
            const unsigned short* vr = &Vt[buf][dsub * 16 + row][0];
            #pragma unroll
            for (int v = 0; v < 8; ++v)
                vf[dsub].u[v] = *(const unsigned int*)(vr + (v >> 2) * 16 + g * 8 + 2 * (v & 3));
        }
        #pragma unroll
        for (int dsub = 0; dsub < 8; ++dsub)
            acc_o[dsub] = __builtin_amdgcn_wmma_f32_16x16x32_bf16(
                false, pf.v, false, vf[dsub].v, (short)0, acc_o[dsub], false, false);

        __syncthreads();                        // buf free for refill
        buf ^= 1;
    }

    // normalize + store bf16 into proj-GEMM A buffer
    #pragma unroll
    for (int v = 0; v < 8; ++v) {
        float inv = 1.0f / lrow[v];
        size_t r = rowQ0 + w * 16 + v + 8 * g;
        unsigned short* dst = out_bf + r * 2048 + h * 128;
        #pragma unroll
        for (int dsub = 0; dsub < 8; ++dsub)
            dst[dsub * 16 + row] = f2bf(acc_o[dsub][v] * inv);
    }
}

// ---------------------------------------------------------------------------
extern "C" void kernel_launch(void* const* d_in, const int* in_sizes, int n_in,
                              void* d_out, int out_size, void* d_ws, size_t ws_size,
                              hipStream_t stream) {
    const float* hidden   = (const float*)d_in[0];
    // d_in[1]: attention_mask (causal, known statically) -- unused
    const float* c_attn_w = (const float*)d_in[2];
    const float* c_attn_b = (const float*)d_in[3];
    const float* c_proj_w = (const float*)d_in[4];
    const float* c_proj_b = (const float*)d_in[5];
    float* out = (float*)d_out;

    const int Bb = 2, S = 2048, E = 2048, Dh = 128, H = 16;
    const int M = Bb * S;            // 4096
    const int Nqkv = E + 2 * Dh;     // 2304

    unsigned short* hid_bf  = (unsigned short*)d_ws;
    unsigned short* wat_t   = hid_bf  + (size_t)M * E;        // [2304][2048]
    unsigned short* wpr_t   = wat_t   + (size_t)Nqkv * E;     // [2048][2048]
    unsigned short* attn_bf = wpr_t   + (size_t)E * E;        // [4096][2048]
    unsigned short* qkv_bf  = attn_bf + (size_t)M * E;        // [4096][2304]
    unsigned short* vt_ws   = qkv_bf  + (size_t)M * Nqkv;     // [2][128][2048]

    // 1) hidden fp32 -> bf16
    f32_to_bf16_kernel<<<(M * E) / 1024, 256, 0, stream>>>(hidden, hid_bf, M * E);
    // 2) weight transposes (-> [N][K] bf16)
    transpose_f32_to_bf16<<<dim3(Nqkv / 32, E / 32), 256, 0, stream>>>(c_attn_w, wat_t, E, Nqkv);
    transpose_f32_to_bf16<<<dim3(E / 32, E / 32), 256, 0, stream>>>(c_proj_w, wpr_t, E, E);
    // 3) QKV projection GEMM -> bf16 qkv (+bias)
    gemm_bf16_wmma<<<dim3(Nqkv / 128, M / 128), 256, 0, stream>>>(
        hid_bf, wat_t, c_attn_b, nullptr, qkv_bf, M, Nqkv, E);
    // 4) V transpose for the PV GEMM B-layout
    transpose_v_bf16<<<dim3(S / 32, 4, Bb), 256, 0, stream>>>(qkv_bf, vt_ws, S);
    // 5) causal MQA flash attention -> bf16 activations
    attn_flash_wmma<<<dim3(S / 64, Bb * H), 128, 0, stream>>>(qkv_bf, vt_ws, attn_bf, S);
    // 6) output projection GEMM -> d_out (fp32 + bias)
    gemm_bf16_wmma<<<dim3(E / 128, M / 128), 256, 0, stream>>>(
        attn_bf, wpr_t, c_proj_b, out, nullptr, M, E, E);
}
